// Decoder_50654844289746
// MI455X (gfx1250) — matrix operand order repaired and checked by reading, not compiled, not run on hardware
//
#include <hip/hip_runtime.h>
#include <hip/hip_bf16.h>
#include <stdint.h>

// ---------------------------------------------------------------------------
// MI455X (gfx1250) implementation.
//  * All GEMMs (projections, FFN, coarse/sim/offset MLPs) -> v_wmma_f32_16x16x32_f16
//  * Attention fused flash-style per wave (no 256MB logits buffer; L2-resident)
//  * wave32 fragment layouts per CDNA5 ISA 7.12.2
//  * All GEMM shapes padded to 16-multiples -> branch-free inner loop:
//    2x global_load_b128 (A) + 16x 2B strided loads (B) + v_wmma per K-step.
// ---------------------------------------------------------------------------

typedef _Float16 half_t;
typedef __attribute__((ext_vector_type(16))) _Float16 v16h;
typedef __attribute__((ext_vector_type(8)))  _Float16 v8h;
typedef __attribute__((ext_vector_type(8)))  float    v8f;

#define DEV __device__ __forceinline__

// D = A(16x32) * B(32x16) + C, f16 inputs, f32 accum
DEV v8f wmma32(v16h a, v16h b, v8f c) {
  return __builtin_amdgcn_wmma_f32_16x16x32_f16(false, a, false, b, (short)0, c, false, false);
}

// A fragment (16x32, row-major source, lda in elements).
// lane<16 : row=lane,    elems 0..7 = K0..7,  8..15 = K16..23
// lane>=16: row=lane-16, elems 0..7 = K8..15, 8..15 = K24..31
DEV v16h load_a16(const half_t* A, int lda, int row0, int k0) {
  int lane = threadIdx.x & 31;
  size_t base = (size_t)(row0 + (lane & 15)) * lda + (size_t)k0 + ((lane & 16) ? 8 : 0);
  v8h a0 = *reinterpret_cast<const v8h*>(A + base);
  v8h a1 = *reinterpret_cast<const v8h*>(A + base + 16);
  v16h f;
#pragma unroll
  for (int i = 0; i < 8; i++) { f[i] = a0[i]; f[8 + i] = a1[i]; }
  return f;
}

// B fragment (32x16) from K-major storage B[K][N] (ldb elements):
// lane L holds col c0+L%16, elems 0..7 = K(8h..8h+7), 8..15 = K(16+8h..23+8h), h=L/16.
DEV v16h load_b16(const half_t* B, int ldb, int k0, int c0) {
  int lane = threadIdx.x & 31;
  const half_t* p = B + (size_t)k0 * ldb + c0 + (lane & 15);
  int kb = (lane & 16) ? 8 : 0;
  v16h f;
#pragma unroll
  for (int i = 0; i < 8; i++) { f[i] = p[(size_t)(kb + i) * ldb]; f[8 + i] = p[(size_t)(kb + 16 + i) * ldb]; }
  return f;
}

// ---------------------------------------------------------------------------
// Generic batched WMMA GEMM:  C[M,N] = A[M,K] * B[K,N]   (B stored K-major)
// Requires N%16==0, K%32==0, ldb%16==0 (enforced by padding on the host side).
// epilogue: bias per-col / per-row, relu, f32 store (opt. transposed for
// (B,C,N) outputs), f16 store.
// ---------------------------------------------------------------------------
#define F_RELU  1
#define F_BIASC 2
#define F_BIASR 4
#define F_TRANS 8

__global__ void k_gemm(const half_t* __restrict__ A, int lda, long long sA,
                       const half_t* __restrict__ B, int ldb, long long sB,
                       float* C32, int ldc, long long sC32,
                       half_t* C16, int ldc16, long long sC16,
                       const float* __restrict__ bias,
                       int M, int N, int K, int flags, int tRows) {
  int z = blockIdx.z;
  A += (size_t)z * sA;
  B += (size_t)z * sB;
  if (C32) C32 += (size_t)z * sC32;
  if (C16) C16 += (size_t)z * sC16;
  int wid  = threadIdx.x >> 5;
  int row0 = blockIdx.y * 32 + (wid >> 2) * 16;
  int col0 = blockIdx.x * 64 + (wid & 3) * 16;
  if (row0 >= M || col0 >= N) return;
  v8f acc = {};
  for (int k = 0; k < K; k += 32) {
    v16h a = load_a16(A, lda, row0, k);
    v16h b = load_b16(B, ldb, k, col0);
    acc = wmma32(a, b, acc);
  }
  int lane = threadIdx.x & 31;
  int n  = col0 + (lane & 15);
  int mb = row0 + ((lane & 16) ? 8 : 0);
#pragma unroll
  for (int i = 0; i < 8; i++) {
    int m = mb + i;
    float v = acc[i];
    if (flags & F_BIASC) v += bias[n];
    if (flags & F_BIASR) v += bias[m];
    if (flags & F_RELU)  v = fmaxf(v, 0.f);
    if (C32) {
      if (flags & F_TRANS)
        C32[((size_t)(m / tRows) * N + n) * tRows + (m % tRows)] = v;
      else
        C32[(size_t)m * ldc + n] = v;
    }
    if (C16) C16[(size_t)m * ldc16 + n] = (half_t)v;
  }
}

// ---------------------------------------------------------------------------
// Fused flash attention. dh = 32 (one WMMA K-step), N = 1024 keys.
// q: f16 token-major [B*1024, 256] (head slice via +h*32, ld 256)
// kT: f16 [B*H, 32, 1024] ; v: f16 token-major ; out: f16 token-major
// One wave = 16 queries; key tiles of 32; P staged through per-wave LDS to
// convert the C-fragment into an A-fragment.
// ---------------------------------------------------------------------------
__global__ void k_flash(const half_t* __restrict__ q, const half_t* __restrict__ kT,
                        const half_t* __restrict__ v, const unsigned char* __restrict__ km,
                        half_t* __restrict__ out) {
  int b = blockIdx.z, h = blockIdx.y;
  int wid = threadIdx.x >> 5, lane = threadIdx.x & 31;
  int q0 = (blockIdx.x * 4 + wid) * 16;
  __shared__ alignas(32) half_t P[4][16 * 32];
  half_t* Pw = P[wid];

  const half_t* Qb = q + (size_t)b * 1024 * 256 + h * 32;
  const half_t* Kb = kT + (size_t)(b * 8 + h) * 32 * 1024;
  const half_t* Vb = v + (size_t)b * 1024 * 256 + h * 32;

  v16h af = load_a16(Qb, 256, q0, 0);  // Q fragment, held across whole loop

  float mrow[8], lrow[8];
  v8f o0 = {}, o1 = {};
#pragma unroll
  for (int i = 0; i < 8; i++) { mrow[i] = -1e30f; lrow[i] = 0.f; }
  const float scale = 0.17677669529663687f;  // 1/sqrt(32)

  for (int j0 = 0; j0 < 1024; j0 += 32) {
    v16h kb0 = load_b16(Kb, 1024, 0, j0);
    v16h kb1 = load_b16(Kb, 1024, 0, j0 + 16);
    v8f zz = {};
    v8f s0 = wmma32(af, kb0, zz);
    v8f s1 = wmma32(af, kb1, zz);
    int n0 = j0 + (lane & 15);
    bool msk0 = km[b * 1024 + n0] != 0;
    bool msk1 = km[b * 1024 + n0 + 16] != 0;
#pragma unroll
    for (int i = 0; i < 8; i++) {
      s0[i] = msk0 ? -1e9f : s0[i] * scale;
      s1[i] = msk1 ? -1e9f : s1[i] * scale;
    }
#pragma unroll
    for (int i = 0; i < 8; i++) {
      float tm = fmaxf(s0[i], s1[i]);
      for (int m = 1; m < 16; m <<= 1) tm = fmaxf(tm, __shfl_xor(tm, m, 32));
      float mn = fmaxf(mrow[i], tm);
      float alpha = __expf(mrow[i] - mn);
      float p0 = __expf(s0[i] - mn);
      float p1 = __expf(s1[i] - mn);
      float ps = p0 + p1;
      for (int m = 1; m < 16; m <<= 1) ps += __shfl_xor(ps, m, 32);
      lrow[i] = lrow[i] * alpha + ps;
      o0[i] *= alpha;
      o1[i] *= alpha;
      mrow[i] = mn;
      int mr = i + ((lane & 16) ? 8 : 0);
      int c  = lane & 15;
      Pw[mr * 32 + c]      = (half_t)p0;
      Pw[mr * 32 + c + 16] = (half_t)p1;
    }
    __asm__ volatile("" ::: "memory");  // keep DS stores before DS loads (HW in-order per wave)
    {
      int r = lane & 15, kb = (lane & 16) ? 8 : 0;
      v8h x0 = *reinterpret_cast<const v8h*>(Pw + r * 32 + kb);
      v8h x1 = *reinterpret_cast<const v8h*>(Pw + r * 32 + kb + 16);
      v16h pf;
#pragma unroll
      for (int i = 0; i < 8; i++) { pf[i] = x0[i]; pf[8 + i] = x1[i]; }
      v16h vb0 = load_b16(Vb, 256, j0, 0);
      v16h vb1 = load_b16(Vb, 256, j0, 16);
      o0 = wmma32(pf, vb0, o0);
      o1 = wmma32(pf, vb1, o1);
    }
    __asm__ volatile("" ::: "memory");
  }
  int n  = lane & 15;
  int mb = q0 + ((lane & 16) ? 8 : 0);
#pragma unroll
  for (int i = 0; i < 8; i++) {
    int m = mb + i;
    float inv = 1.f / lrow[i];
    out[((size_t)b * 1024 + m) * 256 + h * 32 + n]      = (half_t)(o0[i] * inv);
    out[((size_t)b * 1024 + m) * 256 + h * 32 + 16 + n] = (half_t)(o1[i] * inv);
  }
}

// ---------------------------------------------------------------------------
// Elementwise / utility kernels
// ---------------------------------------------------------------------------
// f32 -> f16; trans=0: dst[r*ldo+c]; trans=1: dst[c*ldo+r] (ldo = output ld)
__global__ void k_convert(const float* __restrict__ src, half_t* __restrict__ dst,
                          int R, int C, int trans, int ldo) {
  int i = blockIdx.x * blockDim.x + threadIdx.x;
  if (i >= R * C) return;
  int r = i / C, c = i % C;
  float v = src[i];
  if (trans) dst[(size_t)c * ldo + r] = (half_t)v;
  else       dst[(size_t)r * ldo + c] = (half_t)v;
}

__global__ void k_fill_half(half_t* p, int n) {
  int i = blockIdx.x * blockDim.x + threadIdx.x;
  if (i < n) p[i] = (half_t)0.f;
}

// dst[0..npad): src[i] for i<n else 0
__global__ void k_pad_bias(const float* __restrict__ src, float* __restrict__ dst,
                           int n, int npad) {
  int i = blockIdx.x * blockDim.x + threadIdx.x;
  if (i < npad) dst[i] = (i < n) ? src[i] : 0.f;
}

__global__ void k_addpos(const float* __restrict__ x, const float* __restrict__ p,
                         half_t* __restrict__ o, int n) {
  int i = blockIdx.x * blockDim.x + threadIdx.x;
  if (i < n) o[i] = (half_t)(x[i] + p[i]);
}

// descriptor (B,131,1024) -> f16 (B,128,1024) channel-major and (B,1024,128) token-major
__global__ void k_extract(const float* __restrict__ desc, half_t* __restrict__ cn,
                          half_t* __restrict__ nc) {
  int i = blockIdx.x * blockDim.x + threadIdx.x;
  if (i >= 8 * 128 * 1024) return;
  int n = i & 1023, c = (i >> 10) & 127, b = i >> 17;
  float v = desc[((size_t)b * 131 + c) * 1024 + n];
  cn[i] = (half_t)v;
  nc[((size_t)(b * 1024 + n)) * 128 + c] = (half_t)v;
}

// sinusoidal position embedding, token-major f32 [8192,256]
__global__ void k_pos(const float* __restrict__ desc, float* __restrict__ pos) {
  int p = blockIdx.x * blockDim.x + threadIdx.x;
  if (p >= 8192) return;
  int b = p >> 10, n = p & 1023;
  float xyz[3];
#pragma unroll
  for (int k = 0; k < 3; k++) xyz[k] = desc[((size_t)b * 131 + 128 + k) * 1024 + n];
  float* o = pos + (size_t)p * 256;
  const float L = 9.210340371976184f;  // ln(10000)
  for (int i = 0; i < 3; i++)
    for (int j = 0; j < 42; j++) {
      float f = __expf(-((float)j / 42.f) * L);
      float a = xyz[i] * f;
      o[i * 84 + j]      = __sinf(a);
      o[i * 84 + 42 + j] = __cosf(a);
    }
  o[252] = 0.f; o[253] = 0.f; o[254] = 0.f; o[255] = 0.f;
}

// k token-major [B*1024,256] -> kT [B*H,32,1024]
__global__ void k_ktrans(const half_t* __restrict__ kt, half_t* __restrict__ kT) {
  int i = blockIdx.x * blockDim.x + threadIdx.x;
  if (i >= 8 * 8 * 32 * 1024) return;
  int n = i & 1023, d = (i >> 10) & 31, bh = i >> 15;
  kT[i] = kt[((size_t)((bh >> 3) * 1024 + n)) * 256 + (bh & 7) * 32 + d];
}

// LayerNorm over 256 channels; one wave per token. out = LN(resid+delta)*g+b
__global__ void k_ln(const float* __restrict__ resid, const float* __restrict__ delta,
                     const float* __restrict__ g, const float* __restrict__ bb,
                     float* __restrict__ out32, half_t* __restrict__ out16) {
  int wid = threadIdx.x >> 5, lane = threadIdx.x & 31;
  int t = blockIdx.x * 8 + wid;
  const float* r  = resid + (size_t)t * 256;
  const float* dl = delta + (size_t)t * 256;
  float x[8], s = 0.f;
#pragma unroll
  for (int i = 0; i < 8; i++) { int c = i * 32 + lane; x[i] = r[c] + dl[c]; s += x[i]; }
  for (int m = 1; m < 32; m <<= 1) s += __shfl_xor(s, m, 32);
  float mu = s * (1.f / 256.f), vv = 0.f;
#pragma unroll
  for (int i = 0; i < 8; i++) { float d = x[i] - mu; vv += d * d; }
  for (int m = 1; m < 32; m <<= 1) vv += __shfl_xor(vv, m, 32);
  float rstd = rsqrtf(vv * (1.f / 256.f) + 1e-5f);
#pragma unroll
  for (int i = 0; i < 8; i++) {
    int c = i * 32 + lane;
    float y = (x[i] - mu) * rstd * g[c] + bb[c];
    out32[(size_t)t * 256 + c] = y;
    if (out16) out16[(size_t)t * 256 + c] = (half_t)y;
  }
}

__global__ void k_fill_int(int* p, int n, int v) {
  int i = blockIdx.x * blockDim.x + threadIdx.x;
  if (i < n) p[i] = v;
}

// pair search: one thread per (b,i) row
DEV void gt_point(const float* desc, const float* R, const float* T, int b, int i, float* sg) {
  float sx[3];
#pragma unroll
  for (int k = 0; k < 3; k++) sx[k] = desc[((size_t)b * 131 + 128 + k) * 1024 + i];
#pragma unroll
  for (int a = 0; a < 3; a++)
    sg[a] = R[b * 9 + a * 3 + 0] * sx[0] + R[b * 9 + a * 3 + 1] * sx[1] +
            R[b * 9 + a * 3 + 2] * sx[2] + T[b * 3 + a];
}

__global__ void k_pair_count(const float* __restrict__ sdesc, const float* __restrict__ ddesc,
                             const float* __restrict__ R, const float* __restrict__ T,
                             const unsigned char* __restrict__ sm, const unsigned char* __restrict__ dm,
                             int* __restrict__ rowcnt) {
  int row = blockIdx.x * blockDim.x + threadIdx.x;
  if (row >= 8192) return;
  int b = row >> 10, i = row & 1023;
  int cnt = 0;
  if (!sm[b * 1024 + i]) {
    float sg[3];
    gt_point(sdesc, R, T, b, i, sg);
    const float* dx = ddesc + ((size_t)b * 131 + 128) * 1024;
    for (int j = 0; j < 1024; j++) {
      if (dm[b * 1024 + j]) continue;
      float d0 = sg[0] - dx[j], d1 = sg[1] - dx[1024 + j], d2 = sg[2] - dx[2048 + j];
      if (d0 * d0 + d1 * d1 + d2 * d2 <= 0.09f) cnt++;
    }
  }
  rowcnt[row] = cnt;
}

__global__ void k_scan(const int* __restrict__ cnt, int* __restrict__ off) {
  if (threadIdx.x == 0 && blockIdx.x == 0) {
    int a = 0;
    for (int r = 0; r < 8192; r++) { off[r] = a; a += cnt[r]; }
    off[8192] = a;
  }
}

__global__ void k_pair_write(const float* __restrict__ sdesc, const float* __restrict__ ddesc,
                             const float* __restrict__ R, const float* __restrict__ T,
                             const unsigned char* __restrict__ sm, const unsigned char* __restrict__ dm,
                             const int* __restrict__ off, int* __restrict__ pairIdx) {
  int row = blockIdx.x * blockDim.x + threadIdx.x;
  if (row >= 8192) return;
  int b = row >> 10, i = row & 1023;
  if (sm[b * 1024 + i]) return;
  int slot = off[row];
  float sg[3];
  gt_point(sdesc, R, T, b, i, sg);
  const float* dx = ddesc + ((size_t)b * 131 + 128) * 1024;
  for (int j = 0; j < 1024; j++) {
    if (dm[b * 1024 + j]) continue;
    float d0 = sg[0] - dx[j], d1 = sg[1] - dx[1024 + j], d2 = sg[2] - dx[2048 + j];
    if (d0 * d0 + d1 * d1 + d2 * d2 <= 0.09f) {
      if (slot < 32768) pairIdx[slot] = row * 1024 + j;  // flat (b,i,j)
      slot++;
    }
  }
}

// build concatenated pair features f16 [32768, 512]; swap=0 -> [sf,df], 1 -> [df,sf]
__global__ void k_gather(const int* __restrict__ pairIdx, const float* __restrict__ s32,
                         const float* __restrict__ d32, half_t* __restrict__ cat, int swap) {
  size_t e = (size_t)blockIdx.x * blockDim.x + threadIdx.x;
  if (e >= (size_t)32768 * 512) return;
  int p = (int)(e >> 9), c = (int)(e & 511);
  int idx = pairIdx[p];
  int bi = idx >> 20, rem = idx & 0xFFFFF, si = rem >> 10, di = rem & 1023;
  float v;
  if (!swap) v = (c < 256) ? s32[((size_t)bi * 1024 + si) * 256 + c]
                           : d32[((size_t)bi * 1024 + di) * 256 + (c - 256)];
  else       v = (c < 256) ? d32[((size_t)bi * 1024 + di) * 256 + c]
                           : s32[((size_t)bi * 1024 + si) * 256 + (c - 256)];
  cat[e] = (half_t)v;
}

// residual vs GT offsets, * valid. head is [32768,16] (padded), only cols 0..2 used.
__global__ void k_off_final(const float* __restrict__ head, const int* __restrict__ pairIdx,
                            const int* __restrict__ off, const float* __restrict__ sdesc,
                            const float* __restrict__ ddesc, const float* __restrict__ R,
                            const float* __restrict__ T, float* __restrict__ out, int mode) {
  int p = blockIdx.x * blockDim.x + threadIdx.x;
  if (p >= 32768) return;
  float valid = (p < off[8192]) ? 1.f : 0.f;
  int idx = pairIdx[p];
  int bi = idx >> 20, rem = idx & 0xFFFFF, si = rem >> 10, di = rem & 1023;
  float sp[3], dp[3];
  gt_point(sdesc, R, T, bi, si, sp);
#pragma unroll
  for (int k = 0; k < 3; k++) dp[k] = ddesc[((size_t)bi * 131 + 128 + k) * 1024 + di];
  float g[3];
  if (mode == 0) {  // R^T (dp - sp)
    float v0 = dp[0] - sp[0], v1 = dp[1] - sp[1], v2 = dp[2] - sp[2];
#pragma unroll
    for (int a = 0; a < 3; a++)
      g[a] = R[bi * 9 + 0 * 3 + a] * v0 + R[bi * 9 + 1 * 3 + a] * v1 + R[bi * 9 + 2 * 3 + a] * v2;
  } else {
#pragma unroll
    for (int a = 0; a < 3; a++) g[a] = sp[a] - dp[a];
  }
#pragma unroll
  for (int a = 0; a < 3; a++) out[(size_t)p * 3 + a] = (head[(size_t)p * 16 + a] - g[a]) * valid;
}

// ---------------------------------------------------------------------------
// Host orchestration
// ---------------------------------------------------------------------------
static inline int cdiv(int a, int b) { return (a + b - 1) / b; }

extern "C" void kernel_launch(void* const* d_in, const int* in_sizes, int n_in,
                              void* d_out, int out_size, void* d_ws, size_t ws_size,
                              hipStream_t stream) {
  (void)in_sizes; (void)n_in; (void)out_size; (void)ws_size;
  const int T = 8192;

  auto F = [&](int i) { return (const float*)d_in[i]; };
  const float* srcD = F(0);
  const float* dstD = F(1);
  const unsigned char* smask = (const unsigned char*)d_in[2];
  const unsigned char* dmask = (const unsigned char*)d_in[3];
  const float* gtR = F(4);
  const float* gtT = F(5);

  // Param leaves: JAX pytree order (dict keys sorted) appended after the 6 arrays.
  // coarse{b1,b2,w1,w2}=6..9; layers[l] base 10+26l:
  //   cross{k_b,k_w,o_b,o_w,q_b,q_w,v_b,v_w}=+0..7, ffn_b1=+8, ffn_b2=+9,
  //   ffn_w1=+10, ffn_w2=+11, ln1_b,ln1_g,ln2_b,ln2_g,ln3_b,ln3_g=+12..17,
  //   self{...}=+18..25
  // offset{b1,b2,w1,w2}=62..65; proj{b,w}=66..67; sim{b1,b2,w1,w2}=68..71
  const float* coarse_b1 = F(6);  const float* coarse_b2 = F(7);
  const float* coarse_w1 = F(8);  const float* coarse_w2 = F(9);
  const float* off_b1 = F(62); const float* off_b2 = F(63);
  const float* off_w1 = F(64); const float* off_w2 = F(65);
  const float* proj_b = F(66); const float* proj_w = F(67);
  const float* sim_b1 = F(68); const float* sim_b2 = F(69);
  const float* sim_w1 = F(70); const float* sim_w2 = F(71);

  // workspace bump allocator
  char* wp = (char*)d_ws;
  auto alloc = [&](size_t bytes) -> void* {
    void* p = (void*)wp;
    wp += (bytes + 255) & ~(size_t)255;
    return p;
  };
  auto allocH = [&](size_t n) { return (half_t*)alloc(n * sizeof(half_t)); };
  auto allocF = [&](size_t n) { return (float*)alloc(n * sizeof(float)); };

  // f16 weights (K-major "B" operands; coarse kept as A operand)
  half_t* projWT = allocH(128 * 256);
  half_t* wSelf[2][4], *wCross[2][4];       // q,k,v,o transposed
  for (int l = 0; l < 2; l++)
    for (int j = 0; j < 4; j++) { wSelf[l][j] = allocH(65536); wCross[l][j] = allocH(65536); }
  half_t* ffn1T[2], *ffn2T[2];
  for (int l = 0; l < 2; l++) { ffn1T[l] = allocH(256 * 1024); ffn2T[l] = allocH(1024 * 256); }
  half_t* simW1T = allocH(65536);
  half_t* simW2T = allocH(65536);
  half_t* cw1_16 = allocH(16384);
  half_t* cw2_16 = allocH(16384);
  half_t* offW1T = allocH(512 * 256);
  half_t* offW2T = allocH(256 * 16);     // padded: 3 real cols + 13 zero cols
  float*  offB2p = allocF(16);           // padded bias

  // activations
  half_t* feaS16 = allocH((size_t)8 * 128 * 1024);
  half_t* feaD16 = allocH((size_t)8 * 128 * 1024);
  half_t* projinS = allocH((size_t)T * 128);
  half_t* projinD = allocH((size_t)T * 128);
  float*  posS = allocF((size_t)T * 256);
  float*  posD = allocF((size_t)T * 256);
  float*  s32  = allocF((size_t)T * 256);
  float*  d32  = allocF((size_t)T * 256);
  half_t* qin16 = allocH((size_t)T * 256);
  half_t* kin16 = allocH((size_t)T * 256);
  half_t* vin16 = allocH((size_t)T * 256);
  half_t* qtok = allocH((size_t)T * 256);
  half_t* ktok = allocH((size_t)T * 256);
  half_t* vtok = allocH((size_t)T * 256);
  half_t* kTb  = allocH((size_t)8 * 8 * 32 * 1024);
  half_t* attn16 = allocH((size_t)T * 256);
  float*  oproj32 = allocF((size_t)T * 256);
  half_t* ffnh16 = allocH((size_t)T * 1024);
  float*  s2S32 = allocF((size_t)T * 256);
  float*  s2D32 = allocF((size_t)T * 256);
  half_t* s2S16 = allocH((size_t)T * 256);
  half_t* s2D16 = allocH((size_t)T * 256);
  half_t* coarseh = allocH((size_t)8 * 128 * 1024);
  int* rowcnt = (int*)alloc(8192 * sizeof(int));
  int* rowoff = (int*)alloc(8193 * sizeof(int));
  int* pairIdx = (int*)alloc(32768 * sizeof(int));
  half_t* cat16 = allocH((size_t)32768 * 512);
  half_t* offh16 = allocH((size_t)32768 * 256);
  float* offout = allocF((size_t)32768 * 16);   // padded head output

  float* out = (float*)d_out;
  float* o_sim_s = out;
  float* o_sim_d = out + 2097152;
  float* o_co_s  = out + 4194304;
  float* o_co_d  = out + 5242880;
  float* o_off_s = out + 6291456;
  float* o_off_d = out + 6389760;

  auto cvt = [&](const float* src, half_t* dst, int R, int C, int trans, int ldo) {
    k_convert<<<cdiv(R * C, 256), 256, 0, stream>>>(src, dst, R, C, trans, ldo);
  };
  auto gemm = [&](const half_t* A, int lda, long long sA, const half_t* B, int ldb, long long sB,
                  float* C32, int ldc, long long sC32, half_t* C16, int ldc16, long long sC16,
                  const float* bias, int M, int N, int K, int flags, int batch, int tRows) {
    dim3 g(cdiv(N, 64), cdiv(M, 32), batch);
    k_gemm<<<g, 256, 0, stream>>>(A, lda, sA, B, ldb, sB, C32, ldc, sC32, C16, ldc16, sC16,
                                  bias, M, N, K, flags, tRows);
  };

  // ---- weight conversion (f32 -> f16, K-major) ----
  cvt(proj_w, projWT, 256, 128, 1, 256);
  cvt(sim_w1, simW1T, 256, 256, 1, 256);
  cvt(sim_w2, simW2T, 256, 256, 1, 256);
  cvt(coarse_w1, cw1_16, 128, 128, 0, 128);
  cvt(coarse_w2, cw2_16, 128, 128, 0, 128);
  cvt(off_w1, offW1T, 256, 512, 1, 256);
  k_fill_half<<<cdiv(256 * 16, 256), 256, 0, stream>>>(offW2T, 256 * 16);
  cvt(off_w2, offW2T, 3, 256, 1, 16);   // [3,256] -> [256,16] padded
  k_pad_bias<<<1, 32, 0, stream>>>(off_b2, offB2p, 3, 16);

  struct AttP { const float *qw, *qb, *kw, *kb, *vw, *vb, *ow, *ob; } aSelf[2], aCross[2];
  const float *lnG[2][3], *lnB[2][3], *ffnB1[2], *ffnB2[2];
  for (int l = 0; l < 2; l++) {
    int b0 = 10 + l * 26;
    aCross[l] = { F(b0 + 5), F(b0 + 4), F(b0 + 1), F(b0 + 0), F(b0 + 7), F(b0 + 6), F(b0 + 3), F(b0 + 2) };
    aSelf[l]  = { F(b0 + 23), F(b0 + 22), F(b0 + 19), F(b0 + 18), F(b0 + 25), F(b0 + 24), F(b0 + 21), F(b0 + 20) };
    cvt(aCross[l].qw, wCross[l][0], 256, 256, 1, 256);
    cvt(aCross[l].kw, wCross[l][1], 256, 256, 1, 256);
    cvt(aCross[l].vw, wCross[l][2], 256, 256, 1, 256);
    cvt(aCross[l].ow, wCross[l][3], 256, 256, 1, 256);
    cvt(aSelf[l].qw, wSelf[l][0], 256, 256, 1, 256);
    cvt(aSelf[l].kw, wSelf[l][1], 256, 256, 1, 256);
    cvt(aSelf[l].vw, wSelf[l][2], 256, 256, 1, 256);
    cvt(aSelf[l].ow, wSelf[l][3], 256, 256, 1, 256);
    cvt(F(b0 + 10), ffn1T[l], 1024, 256, 1, 1024);
    cvt(F(b0 + 11), ffn2T[l], 256, 1024, 1, 256);
    ffnB1[l] = F(b0 + 8); ffnB2[l] = F(b0 + 9);
    lnB[l][0] = F(b0 + 12); lnG[l][0] = F(b0 + 13);
    lnB[l][1] = F(b0 + 14); lnG[l][1] = F(b0 + 15);
    lnB[l][2] = F(b0 + 16); lnG[l][2] = F(b0 + 17);
  }

  // ---- inputs: feature extraction, coarse MLP, pos-embed, projection ----
  k_extract<<<cdiv(8 * 128 * 1024, 256), 256, 0, stream>>>(srcD, feaS16, projinS);
  k_extract<<<cdiv(8 * 128 * 1024, 256), 256, 0, stream>>>(dstD, feaD16, projinD);

  // coarse MLP (channel-major form: A = W[128,128], B = fea[128,1024] per batch)
  gemm(cw1_16, 128, 0, feaS16, 1024, 131072, nullptr, 0, 0, coarseh, 1024, 131072,
       coarse_b1, 128, 1024, 128, F_BIASR | F_RELU, 8, 1024);
  gemm(cw2_16, 128, 0, coarseh, 1024, 131072, o_co_s, 1024, 131072, nullptr, 0, 0,
       coarse_b2, 128, 1024, 128, F_BIASR, 8, 1024);
  gemm(cw1_16, 128, 0, feaD16, 1024, 131072, nullptr, 0, 0, coarseh, 1024, 131072,
       coarse_b1, 128, 1024, 128, F_BIASR | F_RELU, 8, 1024);
  gemm(cw2_16, 128, 0, coarseh, 1024, 131072, o_co_d, 1024, 131072, nullptr, 0, 0,
       coarse_b2, 128, 1024, 128, F_BIASR, 8, 1024);

  k_pos<<<cdiv(8192, 256), 256, 0, stream>>>(srcD, posS);
  k_pos<<<cdiv(8192, 256), 256, 0, stream>>>(dstD, posD);

  gemm(projinS, 128, 0, projWT, 256, 0, s32, 256, 0, nullptr, 0, 0, proj_b,
       T, 256, 128, F_BIASC, 1, 1024);
  gemm(projinD, 128, 0, projWT, 256, 0, d32, 256, 0, nullptr, 0, 0, proj_b,
       T, 256, 128, F_BIASC, 1, 1024);

  // one multi-head attention + residual LN
  auto attention = [&](const float* xq, const float* pq, const float* xk, const float* pk,
                       const float* xv, const unsigned char* kmask, half_t* const* W,
                       const AttP& P, const float* resid, const float* lg, const float* lb,
                       float* out32p, half_t* out16p) {
    int n = T * 256;
    k_addpos<<<cdiv(n, 256), 256, 0, stream>>>(xq, pq, qin16, n);
    half_t* kinp = qin16;
    if (xk != xq) { k_addpos<<<cdiv(n, 256), 256, 0, stream>>>(xk, pk, kin16, n); kinp = kin16; }
    k_convert<<<cdiv(n, 256), 256, 0, stream>>>(xv, vin16, T, 256, 0, 256);
    gemm(qin16, 256, 0, W[0], 256, 0, nullptr, 0, 0, qtok, 256, 0, P.qb, T, 256, 256, F_BIASC, 1, 1024);
    gemm(kinp,  256, 0, W[1], 256, 0, nullptr, 0, 0, ktok, 256, 0, P.kb, T, 256, 256, F_BIASC, 1, 1024);
    k_ktrans<<<cdiv(8 * 8 * 32 * 1024, 256), 256, 0, stream>>>(ktok, kTb);
    gemm(vin16, 256, 0, W[2], 256, 0, nullptr, 0, 0, vtok, 256, 0, P.vb, T, 256, 256, F_BIASC, 1, 1024);
    k_flash<<<dim3(16, 8, 8), 128, 0, stream>>>(qtok, kTb, vtok, kmask, attn16);
    gemm(attn16, 256, 0, W[3], 256, 0, oproj32, 256, 0, nullptr, 0, 0, P.ob, T, 256, 256, F_BIASC, 1, 1024);
    k_ln<<<T / 8, 256, 0, stream>>>(resid, oproj32, lg, lb, out32p, out16p);
  };

  // ---- transformer layers ----
  for (int l = 0; l < 2; l++) {
    // self attention (in-place residual update)
    attention(s32, posS, s32, posS, s32, smask, wSelf[l], aSelf[l], s32, lnG[l][0], lnB[l][0], s32, nullptr);
    attention(d32, posD, d32, posD, d32, dmask, wSelf[l], aSelf[l], d32, lnG[l][0], lnB[l][0], d32, nullptr);
    // cross attention -> s2 buffers (both read post-self s32/d32)
    attention(s32, posS, d32, posD, d32, dmask, wCross[l], aCross[l], s32, lnG[l][1], lnB[l][1], s2S32, s2S16);
    attention(d32, posD, s32, posS, s32, smask, wCross[l], aCross[l], d32, lnG[l][1], lnB[l][1], s2D32, s2D16);
    // FFN
    gemm(s2S16, 256, 0, ffn1T[l], 1024, 0, nullptr, 0, 0, ffnh16, 1024, 0, ffnB1[l],
         T, 1024, 256, F_BIASC | F_RELU, 1, 1024);
    gemm(ffnh16, 1024, 0, ffn2T[l], 256, 0, oproj32, 256, 0, nullptr, 0, 0, ffnB2[l],
         T, 256, 1024, F_BIASC, 1, 1024);
    k_ln<<<T / 8, 256, 0, stream>>>(s2S32, oproj32, lnG[l][2], lnB[l][2], s32, nullptr);
    gemm(s2D16, 256, 0, ffn1T[l], 1024, 0, nullptr, 0, 0, ffnh16, 1024, 0, ffnB1[l],
         T, 1024, 256, F_BIASC | F_RELU, 1, 1024);
    gemm(ffnh16, 1024, 0, ffn2T[l], 256, 0, oproj32, 256, 0, nullptr, 0, 0, ffnB2[l],
         T, 256, 1024, F_BIASC, 1, 1024);
    k_ln<<<T / 8, 256, 0, stream>>>(s2D32, oproj32, lnG[l][2], lnB[l][2], d32, nullptr);
  }

  // ---- sim MLP (token-major, transposed final store into (B,256,N)) ----
  k_convert<<<cdiv(T * 256, 256), 256, 0, stream>>>(s32, qin16, T, 256, 0, 256);
  gemm(qin16, 256, 0, simW1T, 256, 0, nullptr, 0, 0, ffnh16, 256, 0, sim_b1,
       T, 256, 256, F_BIASC | F_RELU, 1, 1024);
  gemm(ffnh16, 256, 0, simW2T, 256, 0, o_sim_s, 256, 0, nullptr, 0, 0, sim_b2,
       T, 256, 256, F_BIASC | F_TRANS, 1, 1024);
  k_convert<<<cdiv(T * 256, 256), 256, 0, stream>>>(d32, qin16, T, 256, 0, 256);
  gemm(qin16, 256, 0, simW1T, 256, 0, nullptr, 0, 0, ffnh16, 256, 0, sim_b1,
       T, 256, 256, F_BIASC | F_RELU, 1, 1024);
  gemm(ffnh16, 256, 0, simW2T, 256, 0, o_sim_d, 256, 0, nullptr, 0, 0, sim_b2,
       T, 256, 256, F_BIASC | F_TRANS, 1, 1024);

  // ---- correspondence pairs + offset heads ----
  k_pair_count<<<cdiv(8192, 256), 256, 0, stream>>>(srcD, dstD, gtR, gtT, smask, dmask, rowcnt);
  k_scan<<<1, 32, 0, stream>>>(rowcnt, rowoff);
  k_fill_int<<<cdiv(32768, 256), 256, 0, stream>>>(pairIdx, 32768, 0);
  k_pair_write<<<cdiv(8192, 256), 256, 0, stream>>>(srcD, dstD, gtR, gtT, smask, dmask, rowoff, pairIdx);

  for (int mode = 0; mode < 2; mode++) {
    k_gather<<<cdiv(32768 * 512, 256), 256, 0, stream>>>(pairIdx, s32, d32, cat16, mode);
    gemm(cat16, 512, 0, offW1T, 256, 0, nullptr, 0, 0, offh16, 256, 0, off_b1,
         32768, 256, 512, F_BIASC | F_RELU, 1, 1024);
    gemm(offh16, 256, 0, offW2T, 16, 0, offout, 16, 0, nullptr, 0, 0, offB2p,
         32768, 16, 256, F_BIASC, 1, 1024);
    k_off_final<<<cdiv(32768, 256), 256, 0, stream>>>(offout, pairIdx, rowoff, srcD, dstD, gtR, gtT,
                                                      mode == 0 ? o_off_s : o_off_d, mode);
  }
}
